// SpectralConv1d_59227599012555
// MI455X (gfx1250) — matrix operand order repaired and checked by reading, not compiled, not run on hardware
//
#include <hip/hip_runtime.h>
#include <math.h>

// ---------------------------------------------------------------------------
// SpectralConv1d (DHT based) for MI455X / gfx1250, wave32.
// Three dense GEMMs vs cas matrices -> V_WMMA_F32_16X16X32_BF16 with hi/lo
// (Dekker) split accumulation; global->LDS staging via CDNA5 async-to-LDS
// (ASYNCcnt) with double-buffered LDS, one barrier per K-step.
// ---------------------------------------------------------------------------

typedef __attribute__((ext_vector_type(16))) __bf16 v16bf;
typedef __attribute__((ext_vector_type(8)))  __bf16 v8bf;
typedef __attribute__((ext_vector_type(8)))  float  v8f;
typedef __attribute__((ext_vector_type(4)))  int    v4i;

#define N1     4096   // signal length
#define CB     2048   // D*M = 32*64 GEMM column count
#define KHALF  2049   // N//2 + 1
#define ROWS_P 2112   // padded out-rows (>=2049, multiple of 64)
#define K2P    2080   // padded KHALF  (multiple of 32)
#define TWOPI  6.28318530717958647692f

__device__ __forceinline__ void bf16_split(float v, __bf16* hi, __bf16* lo) {
  __bf16 h = (__bf16)v;
  *hi = h;
  *lo = (__bf16)(v - (float)h);
}

// --- CDNA5 async global->LDS (16B per lane), ASYNCcnt tracked ---------------
__device__ __forceinline__ void async_copy_b128(const __bf16* g, __bf16* l) {
#if __has_builtin(__builtin_amdgcn_global_load_async_to_lds_b128)
  __builtin_amdgcn_global_load_async_to_lds_b128(
      (__attribute__((address_space(1))) v4i*)g,
      (__attribute__((address_space(3))) v4i*)l, 0, 0);
#else
  unsigned laddr = (unsigned)(unsigned long long)
      (__attribute__((address_space(3))) __bf16*)l;
  asm volatile("global_load_async_to_lds_b128 %0, %1, off"
               :: "v"(laddr), "v"((unsigned long long)g) : "memory");
#endif
}

__device__ __forceinline__ void wait_async0() {
#if __has_builtin(__builtin_amdgcn_s_wait_asynccnt)
  __builtin_amdgcn_s_wait_asynccnt(0);
#else
  asm volatile("s_wait_asynccnt 0" ::: "memory");
#endif
}

// ---------------------------------------------------------------------------
// Pack x into K-contiguous (transposed) bf16 hi/lo operands for the DHT GEMM.
// xvT [c][k]  = x[d, :, :] viewed as (4096,64):  x[d*262144 + k*64 + m]
// xvfT[c][k]  = same view of flip(x, axis=-1);  c = d*64 + m, k in [0,4096)
// ---------------------------------------------------------------------------
__global__ __launch_bounds__(256) void pack_x_kernel(
    const float* __restrict__ x,
    __bf16* __restrict__ xvT_hi, __bf16* __restrict__ xvT_lo,
    __bf16* __restrict__ xvfT_hi, __bf16* __restrict__ xvfT_lo)
{
  long idx = (long)blockIdx.x * blockDim.x + threadIdx.x;
  if (idx >= (long)CB * N1) return;
  int k = (int)(idx & (N1 - 1));
  int c = (int)(idx >> 12);
  int d = c >> 6, m = c & 63;
  long dbase = (long)d * 64 * N1;

  float v1 = x[dbase + (long)k * 64 + m];
  int j = k * 64 + m;                 // within-d flat index of the reshape view
  int mo = j >> 12, no = j & (N1 - 1);
  float v2 = x[dbase + (long)mo * N1 + (N1 - 1 - no)];

  __bf16 h, l;
  bf16_split(v1, &h, &l);
  xvT_hi[idx] = h;  xvT_lo[idx] = l;
  bf16_split(v2, &h, &l);
  xvfT_hi[idx] = h; xvfT_lo[idx] = l;
}

// cas[r][k] = cos(2*pi*r*k/n) + sin(2*pi*r*k/n), zero-padded to (rowsP, colsP)
__global__ __launch_bounds__(256) void gen_cas_kernel(
    __bf16* __restrict__ hi, __bf16* __restrict__ lo,
    int rowsP, int colsP, int rowsReal, int colsReal, int nmod)
{
  long idx = (long)blockIdx.x * blockDim.x + threadIdx.x;
  if (idx >= (long)rowsP * colsP) return;
  int k = (int)(idx % colsP);
  int r = (int)(idx / colsP);
  float v = 0.0f;
  if (r < rowsReal && k < colsReal) {
    long rk = (long)r * k % nmod;
    float th = (float)rk * (TWOPI / (float)nmod);
    float s, c;
    sincosf(th, &s, &c);
    v = c + s;
  }
  __bf16 h, l;
  bf16_split(v, &h, &l);
  hi[idx] = h; lo[idx] = l;
}

// ---------------------------------------------------------------------------
// GEMM C[M x N] = A[M x K] (row-major) * B^T where B is stored K-contiguous.
// bf16 hi/lo split, fp32 accumulate:  a*b ~= ah*bh + ah*bl + al*bh.
// Block: 256 threads = 8 waves (2x4), BM=64, BN=128, BK=32.
// Each wave owns a 32x32 output patch = 2x2 WMMA 16x16 tiles.
// Double-buffered LDS fed by async-to-LDS; one barrier per K-step.
// ---------------------------------------------------------------------------
#define BM 64
#define BN 128
#define BK 32

__global__ __launch_bounds__(256) void gemm_bf16x2_nt_kernel(
    const __bf16* __restrict__ Ah, const __bf16* __restrict__ Al,
    const __bf16* __restrict__ Bh, const __bf16* __restrict__ Bl,
    float* __restrict__ C, int K, int lda, int ldb, int ldc)
{
  __shared__ __bf16 Ash[2][BM * BK];
  __shared__ __bf16 Asl[2][BM * BK];
  __shared__ __bf16 Bsh[2][BN * BK];
  __shared__ __bf16 Bsl[2][BN * BK];

  const int tid  = threadIdx.x;
  const int lane = tid & 31;
  const int wave = tid >> 5;
  const int wm0  = (wave >> 2) * 32;   // wave grid 2 (M) x 4 (N)
  const int wn0  = (wave & 3) * 32;
  const long tileM = (long)blockIdx.x * BM;
  const long tileN = (long)blockIdx.y * BN;

  v8f acc[2][2] = {};

  // staging assignments: A row = 64B -> 4 threads/row; B row = 64B -> 2/row
  const int arow = tid >> 2, akoff = (tid & 3) * 8;
  const int brow = tid >> 1, bkoff = (tid & 1) * 16;

  const __bf16* gAh = Ah + (tileM + arow) * (long)lda + akoff;
  const __bf16* gAl = Al + (tileM + arow) * (long)lda + akoff;
  const __bf16* gBh = Bh + (tileN + brow) * (long)ldb + bkoff;
  const __bf16* gBl = Bl + (tileN + brow) * (long)ldb + bkoff;
  const int aoff = arow * BK + akoff;
  const int boff = brow * BK + bkoff;

  const int arl = lane & 15;
  const int kb  = (lane >> 4) * 8;     // unused (0 or 8); A reads use kb2
  const int kb2 = (lane >> 4) * 16;    // B lane K-base (0 or 16)

  const int nk = K / BK;

  // stage tile kt into LDS buffer `buf` (6 x 16B async ops per thread)
  auto stage = [&](int buf, int kt) {
    const int k0 = kt * BK;
    async_copy_b128(gAh + k0, &Ash[buf][aoff]);
    async_copy_b128(gAl + k0, &Asl[buf][aoff]);
    async_copy_b128(gBh + k0, &Bsh[buf][boff]);
    async_copy_b128(gBh + k0 + 8, &Bsh[buf][boff + 8]);
    async_copy_b128(gBl + k0, &Bsl[buf][boff]);
    async_copy_b128(gBl + k0 + 8, &Bsl[buf][boff + 8]);
  };

  stage(0, 0);

  for (int kt = 0; kt < nk; ++kt) {
    const int cur = kt & 1;
    wait_async0();          // this wave's tile-kt data is in LDS
    __syncthreads();        // everyone's tile-kt visible; prev reads of other
                            // buffer complete across all waves
    if (kt + 1 < nk) stage(cur ^ 1, kt + 1);

#pragma unroll
    for (int mi = 0; mi < 2; ++mi) {
      // A fragment 16x32: lane L<16 holds row L, K = 0..15;
      // lane L>=16 holds row L-16, K = 16..31.
      const int abase = (wm0 + mi * 16 + arl) * BK;
      v8bf h0 = *(const v8bf*)(&Ash[cur][abase + kb2]);
      v8bf h1 = *(const v8bf*)(&Ash[cur][abase + kb2 + 8]);
      v8bf l0 = *(const v8bf*)(&Asl[cur][abase + kb2]);
      v8bf l1 = *(const v8bf*)(&Asl[cur][abase + kb2 + 8]);
      v16bf a_hi = __builtin_shufflevector(h0, h1, 0,1,2,3,4,5,6,7,8,9,10,11,12,13,14,15);
      v16bf a_lo = __builtin_shufflevector(l0, l1, 0,1,2,3,4,5,6,7,8,9,10,11,12,13,14,15);
#pragma unroll
      for (int ni = 0; ni < 2; ++ni) {
        // B fragment 32x16: lane L<16 -> col L, K=0..15; lane L>=16 -> K=16..31
        const int bbase = (wn0 + ni * 16 + arl) * BK + kb2;
        v16bf b_hi = *(const v16bf*)(&Bsh[cur][bbase]);
        v16bf b_lo = *(const v16bf*)(&Bsl[cur][bbase]);
        acc[mi][ni] = __builtin_amdgcn_wmma_f32_16x16x32_bf16(
            false, a_hi, false, b_hi, (short)0, acc[mi][ni], false, false);
        acc[mi][ni] = __builtin_amdgcn_wmma_f32_16x16x32_bf16(
            false, a_hi, false, b_lo, (short)0, acc[mi][ni], false, false);
        acc[mi][ni] = __builtin_amdgcn_wmma_f32_16x16x32_bf16(
            false, a_lo, false, b_hi, (short)0, acc[mi][ni], false, false);
      }
    }
  }

  // C/D layout: VGPR v -> lanes 0..15 row M=v, lanes 16..31 row M=v+8
  const int rhalf = (lane >> 4) * 8;
  const int cl = lane & 15;
#pragma unroll
  for (int mi = 0; mi < 2; ++mi) {
#pragma unroll
    for (int ni = 0; ni < 2; ++ni) {
      long r0 = tileM + wm0 + mi * 16 + rhalf;
      long cn = tileN + wn0 + ni * 16 + cl;
#pragma unroll
      for (int v = 0; v < 8; ++v)
        C[(r0 + v) * (long)ldc + cn] = acc[mi][ni][v];
    }
  }
}

// ---------------------------------------------------------------------------
// compl_mul1d:  out[d][o][t] = 0.5 * sum_i  x1[d,i,t]*(w[i,o,t]+w[i,o,nt])
//                                         + x1[d,i,nt]*(w[i,o,t]-w[i,o,nt])
// where nt=(64-t)%64 and x1[d,i,t] = X1[t][d*64+i].  Tiny -> plain VALU.
// ---------------------------------------------------------------------------
__global__ __launch_bounds__(256) void modes_kernel(
    const float* __restrict__ X1, const float* __restrict__ W,
    float* __restrict__ outm)
{
  int idx = blockIdx.x * blockDim.x + threadIdx.x;   // 32*64*64
  if (idx >= 32 * 64 * 64) return;
  int t = idx & 63;
  int o = (idx >> 6) & 63;
  int d = idx >> 12;
  int nt = (64 - t) & 63;
  float s = 0.f;
#pragma unroll 4
  for (int i = 0; i < 64; ++i) {
    float x1t = X1[(long)t  * CB + d * 64 + i];
    float x1n = X1[(long)nt * CB + d * 64 + i];
    float wt  = W[((long)i * 64 + o) * 64 + t];
    float wn  = W[((long)i * 64 + o) * 64 + nt];
    s = fmaf(x1t, wt + wn, fmaf(x1n, wt - wn, s));
  }
  outm[idx] = 0.5f * s;
}

__global__ __launch_bounds__(256) void zero_bf16x2_kernel(
    __bf16* __restrict__ a, __bf16* __restrict__ b, long n)
{
  long idx = (long)blockIdx.x * blockDim.x + threadIdx.x;
  if (idx >= n) return;
  a[idx] = (__bf16)0.0f;
  b[idx] = (__bf16)0.0f;
}

// Scatter out_ht's nonzeros through dht3's reshape view into xv2^T (K-contig).
// out_ht[d,m,n] (n<64) -> within-d flat j=m*KHALF+n -> xv2 row j/64, col d*64+j%64
__global__ __launch_bounds__(256) void scatter_xv2_kernel(
    const float* __restrict__ outm,
    __bf16* __restrict__ xv2T_hi, __bf16* __restrict__ xv2T_lo)
{
  int idx = blockIdx.x * blockDim.x + threadIdx.x;   // 32*64*64
  if (idx >= 32 * 64 * 64) return;
  int n = idx & 63;
  int m = (idx >> 6) & 63;
  int d = idx >> 12;
  float val = outm[idx];                 // outm indexed [d][o=m][t=n] == idx
  int j  = m * KHALF + n;
  int kk = j >> 6;                       // xv2 row (GEMM K index), < 2080
  int mm = j & 63;
  int c  = d * 64 + mm;                  // GEMM column
  __bf16 h, l;
  bf16_split(val, &h, &l);
  xv2T_hi[(long)c * K2P + kk] = h;
  xv2T_lo[(long)c * K2P + kk] = l;
}

// out = (Xi/K) * cos(atan2(X2, X1)) = (Xi/K) * X1 / sqrt(X1^2 + X2^2)
__global__ __launch_bounds__(256) void finalize_kernel(
    const float* __restrict__ X1, const float* __restrict__ X2,
    const float* __restrict__ Xi, float* __restrict__ out)
{
  long idx = (long)blockIdx.x * blockDim.x + threadIdx.x;  // 32*64*2049
  if (idx >= (long)CB * KHALF) return;
  int n  = (int)(idx % KHALF);
  int dm = (int)(idx / KHALF);           // = d*64 + m = GEMM column
  float xh = X1[(long)n * CB + dm];
  float xf = X2[(long)n * CB + dm];
  float xi = Xi[(long)n * CB + dm] * (1.0f / (float)KHALF);
  float r2 = xh * xh + xf * xf;
  float cp = (r2 > 0.0f) ? xh * rsqrtf(r2) : 1.0f;   // atan2(0,0)=0 -> cos=1
  out[idx] = xi * cp;
}

// ---------------------------------------------------------------------------
extern "C" void kernel_launch(void* const* d_in, const int* in_sizes, int n_in,
                              void* d_out, int out_size, void* d_ws, size_t ws_size,
                              hipStream_t stream)
{
  const float* x = (const float*)d_in[0];    // (32, 64, 4096) fp32
  const float* w = (const float*)d_in[1];    // (64, 64, 64)  fp32
  float* out = (float*)d_out;                // (32, 64, 2049) fp32
  char* ws = (char*)d_ws;

  size_t off = 0;
  auto take = [&](size_t bytes) -> char* {
    char* p = ws + off;
    off = (off + bytes + 255) & ~(size_t)255;
    return p;
  };

  __bf16* xvT_hi  = (__bf16*)take((size_t)CB * N1 * 2);
  __bf16* xvT_lo  = (__bf16*)take((size_t)CB * N1 * 2);
  __bf16* xvfT_hi = (__bf16*)take((size_t)CB * N1 * 2);
  __bf16* xvfT_lo = (__bf16*)take((size_t)CB * N1 * 2);
  __bf16* cas1_hi = (__bf16*)take((size_t)ROWS_P * N1 * 2);
  __bf16* cas1_lo = (__bf16*)take((size_t)ROWS_P * N1 * 2);
  __bf16* cas2_hi = (__bf16*)take((size_t)ROWS_P * K2P * 2);
  __bf16* cas2_lo = (__bf16*)take((size_t)ROWS_P * K2P * 2);
  __bf16* xv2T_hi = (__bf16*)take((size_t)CB * K2P * 2);
  __bf16* xv2T_lo = (__bf16*)take((size_t)CB * K2P * 2);
  float*  X1      = (float*)take((size_t)ROWS_P * CB * 4);
  float*  X2      = (float*)take((size_t)ROWS_P * CB * 4);
  float*  Xi      = (float*)take((size_t)ROWS_P * CB * 4);
  float*  outm    = (float*)take((size_t)32 * 64 * 64 * 4);

  const int T = 256;

  // 1) pack x (+flip) into K-contiguous bf16 hi/lo
  {
    long n = (long)CB * N1;
    pack_x_kernel<<<dim3((unsigned)((n + T - 1) / T)), dim3(T), 0, stream>>>(
        x, xvT_hi, xvT_lo, xvfT_hi, xvfT_lo);
  }
  // 2) cas matrices
  {
    long n = (long)ROWS_P * N1;
    gen_cas_kernel<<<dim3((unsigned)((n + T - 1) / T)), dim3(T), 0, stream>>>(
        cas1_hi, cas1_lo, ROWS_P, N1, KHALF, N1, N1);
    long n2 = (long)ROWS_P * K2P;
    gen_cas_kernel<<<dim3((unsigned)((n2 + T - 1) / T)), dim3(T), 0, stream>>>(
        cas2_hi, cas2_lo, ROWS_P, K2P, KHALF, KHALF, KHALF);
  }
  // 3) DHT GEMMs: X1 = cas1 @ xv , X2 = cas1 @ flip-view
  {
    dim3 grid(ROWS_P / BM, CB / BN);
    gemm_bf16x2_nt_kernel<<<grid, dim3(T), 0, stream>>>(
        cas1_hi, cas1_lo, xvT_hi, xvT_lo, X1, N1, N1, N1, CB);
    gemm_bf16x2_nt_kernel<<<grid, dim3(T), 0, stream>>>(
        cas1_hi, cas1_lo, xvfT_hi, xvfT_lo, X2, N1, N1, N1, CB);
  }
  // 4) mode mixing (tiny)
  modes_kernel<<<dim3(32 * 64 * 64 / T), dim3(T), 0, stream>>>(X1, w, outm);
  // 5) build xv2^T (zero + sparse scatter through the reshape view)
  {
    long n = (long)CB * K2P;
    zero_bf16x2_kernel<<<dim3((unsigned)((n + T - 1) / T)), dim3(T), 0, stream>>>(
        xv2T_hi, xv2T_lo, n);
    scatter_xv2_kernel<<<dim3(32 * 64 * 64 / T), dim3(T), 0, stream>>>(
        outm, xv2T_hi, xv2T_lo);
  }
  // 6) inverse DHT GEMM: Xi = cas2 @ xv2
  {
    dim3 grid(ROWS_P / BM, CB / BN);
    gemm_bf16x2_nt_kernel<<<grid, dim3(T), 0, stream>>>(
        cas2_hi, cas2_lo, xv2T_hi, xv2T_lo, Xi, K2P, K2P, K2P, CB);
  }
  // 7) fused epilogue: xi/K * cos(atan2(x_ht_flip, x_ht))
  {
    long n = (long)CB * KHALF;
    finalize_kernel<<<dim3((unsigned)((n + T - 1) / T)), dim3(T), 0, stream>>>(
        X1, X2, Xi, out);
  }
}
